// MultiBoxLoss_tf_target_balance_32203664786115
// MI455X (gfx1250) — compile-verified, matrix operand order checked by reading, not run on hardware
//
#include <hip/hip_runtime.h>

typedef __attribute__((ext_vector_type(16))) _Float16 v16h;
typedef __attribute__((ext_vector_type(8)))  float    v8f;

#define TPB 256
#define MAXC 20

// ---------------------------------------------------------------- K0: init
__global__ __launch_bounds__(TPB) void k_init(unsigned long long* bestPrior, int* ovr,
                                              float* lLp, float* lCp, float* lBp,
                                              int* tot, int BP, int BN, int nPart, int nB2) {
    int gid = blockIdx.x * TPB + threadIdx.x;
    if (gid < BP)   ovr[gid] = -1;
    if (gid < BN)   bestPrior[gid] = 0ull;
    if (gid < nPart){ lLp[gid] = 0.f; lCp[gid] = 0.f; }
    if (gid < nB2)  lBp[gid] = 0.f;
    if (gid == 0)   *tot = 0;
}

// ---------------------------------------------------------------- K1: match (IoU)
__global__ __launch_bounds__(TPB) void k_match(const float* __restrict__ priors,
                                               const float* __restrict__ targets,
                                               int P, int NOBJ,
                                               unsigned long long* __restrict__ bestPrior,
                                               float* __restrict__ btOv, int* __restrict__ btIdx) {
    __shared__ float sT[64 * 5];
    __shared__ unsigned long long sPk[64];
    int b   = blockIdx.y;
    int tid = threadIdx.x;
    for (int i = tid; i < NOBJ * 5; i += TPB) sT[i] = targets[(size_t)b * NOBJ * 5 + i];
    for (int i = tid; i < NOBJ;     i += TPB) sPk[i] = 0ull;
    __syncthreads();

    int p = blockIdx.x * TPB + tid;
    if (p < P) {
        float4 prv = *reinterpret_cast<const float4*>(priors + (size_t)p * 4);
        float px1 = prv.x - 0.5f * prv.z, py1 = prv.y - 0.5f * prv.w;
        float px2 = prv.x + 0.5f * prv.z, py2 = prv.y + 0.5f * prv.w;
        float areaB = prv.z * prv.w;
        float bestOv = -1.0f; int bestT = 0;
        for (int t = 0; t < NOBJ; ++t) {
            float tx1 = sT[t*5+0], ty1 = sT[t*5+1], tx2 = sT[t*5+2], ty2 = sT[t*5+3];
            float ix = fminf(px2, tx2) - fmaxf(px1, tx1); ix = fmaxf(ix, 0.f);
            float iy = fminf(py2, ty2) - fmaxf(py1, ty1); iy = fmaxf(iy, 0.f);
            float inter = ix * iy;
            float areaA = (tx2 - tx1) * (ty2 - ty1);
            float iou = inter / (areaA + areaB - inter);
            if (iou > bestOv) { bestOv = iou; bestT = t; }
            unsigned long long pk = ((unsigned long long)__float_as_uint(iou) << 32)
                                  | (unsigned long long)(0xFFFFFFFFu - (unsigned)p);
            atomicMax(&sPk[t], pk);
        }
        btOv[(size_t)b * P + p]  = bestOv;
        btIdx[(size_t)b * P + p] = bestT;
    }
    __syncthreads();
    for (int i = tid; i < NOBJ; i += TPB)
        atomicMax(&bestPrior[(size_t)b * NOBJ + i], sPk[i]);
}

// ---------------------------------------------------------------- K2: override scatter
__global__ __launch_bounds__(TPB) void k_override(const unsigned long long* __restrict__ bestPrior,
                                                  int* __restrict__ ovr, int B, int P, int NOBJ) {
    int id = blockIdx.x * TPB + threadIdx.x;
    if (id >= B * NOBJ) return;
    int b = id / NOBJ, t = id % NOBJ;
    unsigned long long pk = bestPrior[id];
    unsigned pr = 0xFFFFFFFFu - (unsigned)(pk & 0xFFFFFFFFull);
    if (pr < (unsigned)P) atomicMax(&ovr[(size_t)b * P + pr], t);
}

// ---------------------------------------------------------------- K3: fused per-prior losses
__global__ __launch_bounds__(TPB) void k_perprior(const float* __restrict__ loc,
                                                  const float* __restrict__ conf,
                                                  const float* __restrict__ binc,
                                                  const float* __restrict__ priors,
                                                  const float* __restrict__ targets,
                                                  const int* __restrict__ ovr,
                                                  const float* __restrict__ btOv,
                                                  const int* __restrict__ btIdx,
                                                  int* __restrict__ confT, float* __restrict__ ceBin,
                                                  float* __restrict__ lbin, float* __restrict__ lmul,
                                                  float* __restrict__ posOut, float* __restrict__ posWS,
                                                  float* __restrict__ lossLpart, float* __restrict__ lossCpart,
                                                  int B, int P, int NOBJ, int Cm1) {
    int gid = blockIdx.x * TPB + threadIdx.x;
    float myL = 0.f, myC = 0.f;
    if (gid < B * P) {
        int p = gid % P;
        int idx = btIdx[gid]; float ov = btOv[gid];
        int o = ovr[gid];
        if (o >= 0) { idx = o; ov = 2.0f; }
        int b = gid / P;
        const float* tr = targets + ((size_t)b * NOBJ + idx) * 5;
        float mx1 = tr[0], my1 = tr[1], mx2 = tr[2], my2 = tr[3];
        int ct = (int)tr[4] + 1;
        if (ov < 0.5f) ct = 0;
        bool pos = ct > 0;

        float4 prv = *reinterpret_cast<const float4*>(priors + (size_t)p * 4);
        float g[4];
        g[0] = ((mx1 + mx2) * 0.5f - prv.x) / (0.1f * prv.z);
        g[1] = ((my1 + my2) * 0.5f - prv.y) / (0.1f * prv.w);
        g[2] = logf((mx2 - mx1) / prv.z) * 5.0f;
        g[3] = logf((my2 - my1) / prv.w) * 5.0f;
        float4 ldv = *reinterpret_cast<const float4*>(loc + (size_t)gid * 4);
        float ld[4] = {ldv.x, ldv.y, ldv.z, ldv.w};
        float sl1 = 0.f;
        #pragma unroll
        for (int j = 0; j < 4; ++j) {
            float d = fabsf(ld[j] - g[j]);
            sl1 += (d < 1.f) ? 0.5f * d * d : d - 0.5f;
        }
        if (pos) myL = sl1;

        float2 bcv = *reinterpret_cast<const float2*>(binc + (size_t)gid * 2);
        float b0 = bcv.x, b1 = bcv.y;
        float mb = fmaxf(b0, b1);
        float lseB = mb + logf(expf(b0 - mb) + expf(b1 - mb));
        float ceB = lseB - (pos ? b1 : b0);

        const float* cd = conf + (size_t)gid * Cm1;
        __builtin_prefetch(cd + (size_t)64 * Cm1, 0, 1);   // gfx1250 global_prefetch_b8
        float cv[MAXC];
        int NC = (Cm1 < MAXC) ? Cm1 : MAXC;
        if (Cm1 == MAXC) {
            const float4* c4 = reinterpret_cast<const float4*>(cd);  // 80B row, 16B aligned
            #pragma unroll
            for (int j = 0; j < MAXC / 4; ++j) {
                float4 v = c4[j];
                cv[4*j+0] = v.x; cv[4*j+1] = v.y; cv[4*j+2] = v.z; cv[4*j+3] = v.w;
            }
        } else {
            for (int j = 0; j < NC; ++j) cv[j] = cd[j];
        }
        float mc = cv[0];
        for (int j = 1; j < NC; ++j) mc = fmaxf(mc, cv[j]);
        float se = 0.f;
        for (int j = 0; j < NC; ++j) se += expf(cv[j] - mc);
        float lseC = mc + logf(se);
        float P0 = b0 + lseC;
        float mm = fmaxf(P0, mc + b1);
        float sm = expf(P0 - mm);
        for (int j = 0; j < NC; ++j) sm += expf(cv[j] + b1 - mm);
        float lseM = mm + logf(sm);
        float mg = (ct == 0) ? P0 : (cv[ct - 1] + b1);
        float ceM = lseM - mg;
        if (pos) myC = ceM;

        float pf = pos ? 1.f : 0.f;
        confT[gid]  = ct;
        ceBin[gid]  = ceB;
        lbin[gid]   = pos ? 0.f : ceB;
        lmul[gid]   = pos ? 0.f : ceM;
        posOut[gid] = pf;      // output buffer (d_out+3, unaligned for b128)
        posWS[gid]  = pf;      // 16B-aligned workspace copy for WMMA kernel
    }
    __shared__ float rL[TPB], rC[TPB];
    rL[threadIdx.x] = myL; rC[threadIdx.x] = myC;
    __syncthreads();
    for (int off = TPB / 2; off > 0; off >>= 1) {
        if (threadIdx.x < off) { rL[threadIdx.x] += rL[threadIdx.x + off]; rC[threadIdx.x] += rC[threadIdx.x + off]; }
        __syncthreads();
    }
    if (threadIdx.x == 0) { lossLpart[blockIdx.x] = rL[0]; lossCpart[blockIdx.x] = rC[0]; }
}

// ---------------------------------------------------------------- K4: WMMA pos-count
// Counts 0/1 flags per batch row via v_wmma_f32_16x16x32_f16 with B = ones.
// Exact: 0/1 products, <=32-wide f16 dots, f32 accumulation.
// Main body is bounds-check-free (multiples of 512) so each lane issues four
// global_load_b128 with EXEC all-1s at every WMMA; the <512 tail is plain VALU.
__global__ __launch_bounds__(32) void k_countpos(const float* __restrict__ posMask, int P,
                                                 int* __restrict__ numPos, int* __restrict__ numNeg,
                                                 int* __restrict__ numPosTotal) {
    int b = blockIdx.x;
    int lane = threadIdx.x;
    const float* row = posMask + (size_t)b * P;
    v16h ones;
    #pragma unroll
    for (int j = 0; j < 16; ++j) ones[j] = (_Float16)1.0f;
    v8f c = {};
    int Pmain = P & ~511;
    for (int base = 0; base < Pmain; base += 512) {
        const float4* q = reinterpret_cast<const float4*>(row + base + lane * 16);
        float4 f0 = q[0], f1 = q[1], f2 = q[2], f3 = q[3];
        v16h a;
        a[0]  = (_Float16)f0.x; a[1]  = (_Float16)f0.y; a[2]  = (_Float16)f0.z; a[3]  = (_Float16)f0.w;
        a[4]  = (_Float16)f1.x; a[5]  = (_Float16)f1.y; a[6]  = (_Float16)f1.z; a[7]  = (_Float16)f1.w;
        a[8]  = (_Float16)f2.x; a[9]  = (_Float16)f2.y; a[10] = (_Float16)f2.z; a[11] = (_Float16)f2.w;
        a[12] = (_Float16)f3.x; a[13] = (_Float16)f3.y; a[14] = (_Float16)f3.z; a[15] = (_Float16)f3.w;
        c = __builtin_amdgcn_wmma_f32_16x16x32_f16(false, a, false, ones, (short)0, c, false, false);
    }
    float s = 0.f;
    #pragma unroll
    for (int j = 0; j < 8; ++j) s += c[j];
    // C[m][n] = rowsum(m) for every n; lane0 carries rows 0..7, lane16 rows 8..15
    float t0  = __shfl(s, 0, 32);
    float t16 = __shfl(s, 16, 32);
    float ts = 0.f;
    for (int i = Pmain + lane; i < P; i += 32) ts += row[i];
    #pragma unroll
    for (int off = 16; off > 0; off >>= 1) ts += __shfl_xor(ts, off, 32);
    if (lane == 0) {
        int np = (int)(t0 + t16 + ts + 0.5f);
        numPos[b] = np;
        int nn = 3 * np; if (nn > P - 1) nn = P - 1;
        numNeg[b] = nn;
        atomicAdd(numPosTotal, np);
    }
}

// ---------------------------------------------------------------- K5: radix-select (top num_neg threshold)
__global__ __launch_bounds__(TPB) void k_select(const float* __restrict__ lbin,
                                                const float* __restrict__ lmul,
                                                const int* __restrict__ numNeg,
                                                unsigned* __restrict__ pref, int* __restrict__ kRem,
                                                int P, int pass) {
    __shared__ unsigned hist[256];
    int id = blockIdx.x; int b = id >> 1; int arr = id & 1;
    int tid = threadIdx.x;
    if (pass == 0 && tid == 0) {
        int k = numNeg[b];
        kRem[id] = (k > 0) ? k : 0;
        pref[id] = (k > 0) ? 0u : 0xFFFFFFFFu;   // sentinel: selects nothing
    }
    __syncthreads();
    int k = kRem[id];
    unsigned pf = pref[id];
    if (k <= 0) return;
    hist[tid] = 0;
    __syncthreads();
    const float* L = arr ? lmul : lbin;
    size_t row = (size_t)b * P;
    int shP = 32 - 8 * pass;   // compare top 8*pass bits (guarded for pass==0)
    int shB = 24 - 8 * pass;
    for (int i = tid; i < P; i += TPB) {
        unsigned bits = __float_as_uint(L[row + i]);   // losses >= 0 -> order-preserving
        bool ok = (pass == 0) || ((bits >> shP) == (pf >> shP));
        if (ok) atomicAdd(&hist[(bits >> shB) & 0xFFu], 1u);
    }
    __syncthreads();
    if (tid == 0) {
        unsigned c = 0; int q;
        for (q = 255; q >= 0; --q) {
            if (c + hist[q] >= (unsigned)k) break;
            c += hist[q];
        }
        if (q < 0) q = 0;
        pref[id] = pf | ((unsigned)q << shB);
        kRem[id] = k - (int)c;
    }
}

// ---------------------------------------------------------------- K6: ordered tie-break mask + loss_b
__global__ __launch_bounds__(TPB) void k_mask(const float* __restrict__ lbin,
                                              const float* __restrict__ lmul,
                                              const float* __restrict__ ceBin,
                                              const int* __restrict__ confT,
                                              const float* __restrict__ posM,
                                              const unsigned* __restrict__ pref,
                                              const int* __restrict__ kRem,
                                              float* __restrict__ negBin, float* __restrict__ negMul,
                                              float* __restrict__ lossBpart, int P) {
    __shared__ int sc[TPB];
    __shared__ int sbase;
    __shared__ float red[TPB];
    int id = blockIdx.x; int b = id >> 1; int arr = id & 1;
    int tid = threadIdx.x;
    unsigned T = pref[id]; int k = kRem[id];
    const float* L = arr ? lmul : lbin;
    float* outM = arr ? negMul : negBin;
    size_t row = (size_t)b * P;
    if (tid == 0) sbase = 0;
    __syncthreads();
    float accB = 0.f;
    for (int base = 0; base < P; base += TPB) {
        int i = base + tid;
        int eq = 0, sel = 0;
        if (i < P) {
            unsigned bits = __float_as_uint(L[row + i]);
            if (bits > T) sel = 1; else if (bits == T) eq = 1;
        }
        sc[tid] = eq;
        __syncthreads();
        for (int off = 1; off < TPB; off <<= 1) {   // Hillis-Steele inclusive scan
            int v = (tid >= off) ? sc[tid - off] : 0;
            __syncthreads();
            sc[tid] += v;
            __syncthreads();
        }
        int excl = sc[tid] - eq;
        if (eq && (sbase + excl) < k) sel = 1;      // stable tie-break by index
        int tot = sc[TPB - 1];
        __syncthreads();
        if (tid == 0) sbase += tot;
        if (i < P) {
            outM[row + i] = sel ? 1.f : 0.f;
            if (arr == 0) {
                bool pos = posM[row + i] > 0.5f;
                if (pos || sel) {
                    float w = (confT[row + i] == 0) ? 3.f : 1.f;
                    accB += w * ceBin[row + i];
                }
            }
        }
        __syncthreads();
    }
    red[tid] = accB;
    __syncthreads();
    for (int off = TPB / 2; off > 0; off >>= 1) {
        if (tid < off) red[tid] += red[tid + off];
        __syncthreads();
    }
    if (tid == 0) lossBpart[id] = (arr == 0) ? red[0] : 0.f;
}

// ---------------------------------------------------------------- K7: deterministic finalize
__global__ void k_final(const float* __restrict__ lLp, const float* __restrict__ lCp,
                        const float* __restrict__ lBp, int nPartLC, int nPartB,
                        const int* __restrict__ numPosTotal, float* __restrict__ out) {
    if (blockIdx.x == 0 && threadIdx.x == 0) {
        float ll = 0.f, lc = 0.f, lb = 0.f;
        for (int i = 0; i < nPartLC; ++i) { ll += lLp[i]; lc += lCp[i]; }
        for (int i = 0; i < nPartB; ++i)  lb += lBp[i];
        int npt = *numPosTotal;
        float N = (float)((npt > 1) ? npt : 1);
        out[0] = ll / N;
        out[1] = lc / N;
        out[2] = lb / N;
    }
}

// ---------------------------------------------------------------- launch
extern "C" void kernel_launch(void* const* d_in, const int* in_sizes, int n_in,
                              void* d_out, int out_size, void* d_ws, size_t ws_size,
                              hipStream_t stream) {
    const float* loc     = (const float*)d_in[0];
    const float* conf    = (const float*)d_in[1];
    const float* binc    = (const float*)d_in[2];
    const float* priors  = (const float*)d_in[3];
    const float* targets = (const float*)d_in[4];

    int P    = in_sizes[3] / 4;
    int BP   = in_sizes[0] / 4;
    int B    = BP / P;
    int NOBJ = in_sizes[4] / (B * 5);
    int Cm1  = in_sizes[1] / BP;

    float* out = (float*)d_out;
    float* posOut = out + 3;
    float* negBin = out + 3 + (size_t)BP;
    float* negMul = out + 3 + (size_t)2 * BP;

    char* w = (char*)d_ws;
    auto carve = [&](size_t bytes) -> char* {
        char* r = w;
        w += (bytes + 15) & ~(size_t)15;
        return r;
    };
    int nPartLC = (BP + TPB - 1) / TPB;
    unsigned long long* bestPrior = (unsigned long long*)carve((size_t)B * NOBJ * 8);
    int*      ovr    = (int*)     carve((size_t)BP * 4);
    float*    btOv   = (float*)   carve((size_t)BP * 4);
    int*      btIdx  = (int*)     carve((size_t)BP * 4);
    int*      confT  = (int*)     carve((size_t)BP * 4);
    float*    ceBin  = (float*)   carve((size_t)BP * 4);
    float*    lbin   = (float*)   carve((size_t)BP * 4);
    float*    lmul   = (float*)   carve((size_t)BP * 4);
    float*    posWS  = (float*)   carve((size_t)BP * 4);   // 16B-aligned pos mask
    int*      numPos = (int*)     carve((size_t)B * 4);
    int*      numNeg = (int*)     carve((size_t)B * 4);
    unsigned* pref   = (unsigned*)carve((size_t)2 * B * 4);
    int*      kRem   = (int*)     carve((size_t)2 * B * 4);
    float*    lLp    = (float*)   carve((size_t)nPartLC * 4);
    float*    lCp    = (float*)   carve((size_t)nPartLC * 4);
    float*    lBp    = (float*)   carve((size_t)2 * B * 4);
    int*      npTot  = (int*)     carve(4);

    k_init<<<(BP + TPB - 1) / TPB, TPB, 0, stream>>>(bestPrior, ovr, lLp, lCp, lBp, npTot,
                                                     BP, B * NOBJ, nPartLC, 2 * B);
    dim3 g1((P + TPB - 1) / TPB, B);
    k_match<<<g1, TPB, 0, stream>>>(priors, targets, P, NOBJ, bestPrior, btOv, btIdx);
    k_override<<<(B * NOBJ + TPB - 1) / TPB, TPB, 0, stream>>>(bestPrior, ovr, B, P, NOBJ);
    k_perprior<<<nPartLC, TPB, 0, stream>>>(loc, conf, binc, priors, targets, ovr, btOv, btIdx,
                                            confT, ceBin, lbin, lmul, posOut, posWS, lLp, lCp,
                                            B, P, NOBJ, Cm1);
    k_countpos<<<B, 32, 0, stream>>>(posWS, P, numPos, numNeg, npTot);
    for (int pass = 0; pass < 4; ++pass)
        k_select<<<2 * B, TPB, 0, stream>>>(lbin, lmul, numNeg, pref, kRem, P, pass);
    k_mask<<<2 * B, TPB, 0, stream>>>(lbin, lmul, ceBin, confT, posWS, pref, kRem,
                                      negBin, negMul, lBp, P);
    k_final<<<1, 32, 0, stream>>>(lLp, lCp, lBp, nPartLC, 2 * B, npTot, out);
}
